// FlashAttention_9131100471595
// MI455X (gfx1250) — compile-verified, matrix operand order checked by reading, not run on hardware
//
#include <hip/hip_runtime.h>
#include <hip/hip_bf16.h>

// ---------------------------------------------------------------------------
// Causal multi-head attention forward for MI455X (gfx1250, wave32, WMMA).
// Pipeline: f32->bf16 convert, 3x bf16 GEMM (Q,K,V proj), flash attention
// (v_wmma_f32_16x16x32_bf16 for QK^T and PV, online softmax), final GEMM
// (O @ Wo) written in fp32.
// GEMM: 128x128 block tile, 8 waves of 64x32, LDS double-buffered, A tile
// staged with GLOBAL_LOAD_ASYNC_TO_LDS_B128 (ASYNCcnt), B tile transposed
// via packed b32 LDS stores overlapped with WMMA. Attention stages K via
// async-to-LDS and prefetches the next K/V tile (global_prefetch_b8).
// Shapes fixed per reference: B=2, S=2048, D=2048, H=16, dk=128.
// Workspace: ~117 MB of bf16 intermediates.
// ---------------------------------------------------------------------------

typedef __attribute__((ext_vector_type(4)))  __bf16 v4bf;
typedef __attribute__((ext_vector_type(8)))  __bf16 v8bf;
typedef __attribute__((ext_vector_type(16))) __bf16 v16bf;
typedef __attribute__((ext_vector_type(8)))  float  v8f;

static __device__ __forceinline__ v16bf cat16(v8bf lo, v8bf hi) {
  return __builtin_shufflevector(lo, hi, 0,1,2,3,4,5,6,7,8,9,10,11,12,13,14,15);
}
static __device__ __forceinline__ v8f wmma_bf16(v16bf a, v16bf b, v8f c) {
  // D = A(16x32) * B(32x16) + C, fp32 accumulate
  return __builtin_amdgcn_wmma_f32_16x16x32_bf16(false, a, false, b,
                                                 (short)0, c, false, false);
}
static __device__ __forceinline__ unsigned pack2(__bf16 lo, __bf16 hi) {
  unsigned l = __builtin_bit_cast(unsigned short, lo);
  unsigned h = __builtin_bit_cast(unsigned short, hi);
  return l | (h << 16);
}

// Per-lane async DMA: 16 bytes global -> LDS, tracked by ASYNCcnt.
// LDS operand is the low 32 bits of the flat shared address (ISA 10.2:
// the LDS aperture maps addr[31:0] to the LDS byte address).
static __device__ __forceinline__ void async_copy_b128(const __bf16* gsrc,
                                                       __bf16* ldst) {
  unsigned lds = (unsigned)(size_t)ldst;
  unsigned long long ga = (unsigned long long)(size_t)gsrc;
  asm volatile("global_load_async_to_lds_b128 %0, %1, off"
               :: "v"(lds), "v"(ga) : "memory");
}
static __device__ __forceinline__ void wait_async() {
  asm volatile("s_wait_asynccnt 0x0" ::: "memory");
}

// ---------------------------------------------------------------------------
__global__ __launch_bounds__(256) void f32_to_bf16_kernel(
    const float4* __restrict__ in, __bf16* __restrict__ out, int n4) {
  int i = blockIdx.x * blockDim.x + threadIdx.x;
  int stride = gridDim.x * blockDim.x;
  for (; i < n4; i += stride) {
    float4 v = in[i];
    v4bf o = {(__bf16)v.x, (__bf16)v.y, (__bf16)v.z, (__bf16)v.w};
    *(v4bf*)(out + (size_t)i * 4) = o;
  }
}

// ---------------------------------------------------------------------------
// C[MxN] = A[MxK] * B[KxN], all row-major. Block: 256 thr (8 waves),
// block tile 128x128, wave tile 64x32 (4x2 WMMA frags), K stepped by 32,
// double-buffered LDS. OUT_F32 selects fp32 (Cf) vs bf16 (Cb) output.
template <bool OUT_F32>
__global__ __launch_bounds__(256) void gemm_bf16_kernel(
    const __bf16* __restrict__ A, const __bf16* __restrict__ Bm,
    float* __restrict__ Cf, __bf16* __restrict__ Cb,
    int M, int N, int K) {
  __shared__ __bf16 sA[2][128][32];   // [buf][m][k]
  __shared__ __bf16 sB[2][128][32];   // [buf][n][k] (transposed)

  const int tid  = threadIdx.x;
  const int lane = tid & 31;
  const int wave = tid >> 5;
  const int r = lane & 15;          // row/col selector within 16x16 tile
  const int g = lane >> 4;          // lane half-group (K sub-pattern)
  const int m0 = blockIdx.y * 128;
  const int n0 = blockIdx.x * 128;
  const int wm = (wave & 1) * 64;   // wave M offset: 0,64
  const int wn = (wave >> 1) * 32;  // wave N offset: 0,32,64,96

  // Staging coordinates
  const int arow = tid >> 1;        // 0..127
  const int acol = (tid & 1) * 16;  // 0,16
  const int kp   = (tid >> 4) * 2;  // even k: 0..30
  const int nb   = (tid & 15) * 8;  // 0..120

  v8f acc[4][2] = {};
  v8bf b0, b1;

  // Preload tile 0: A via async DMA, B via registers (transposed store)
  {
    const __bf16* gp = A + (size_t)(m0 + arow) * K + acol;
    async_copy_b128(gp,     &sA[0][arow][acol]);
    async_copy_b128(gp + 8, &sA[0][arow][acol + 8]);
    const __bf16* bp = Bm + (size_t)kp * N + n0 + nb;
    b0 = *(const v8bf*)bp;
    b1 = *(const v8bf*)(bp + N);
#pragma unroll
    for (int j = 0; j < 8; ++j)
      *(unsigned*)&sB[0][nb + j][kp] = pack2(b0[j], b1[j]);
  }
  wait_async();
  __syncthreads();

  for (int kk = 0; kk < K; kk += 32) {
    const int cur   = (kk >> 5) & 1;
    const int nxt   = cur ^ 1;
    const bool more = (kk + 32) < K;

    // Kick off next tile: A straight into LDS (async), B into registers.
    // Both overlap with the WMMAs below; buffer nxt is free (barrier above).
    if (more) {
      const __bf16* gp = A + (size_t)(m0 + arow) * K + (kk + 32) + acol;
      async_copy_b128(gp,     &sA[nxt][arow][acol]);
      async_copy_b128(gp + 8, &sA[nxt][arow][acol + 8]);
      const __bf16* bp = Bm + (size_t)(kk + 32 + kp) * N + n0 + nb;
      b0 = *(const v8bf*)bp;
      b1 = *(const v8bf*)(bp + N);
    }

    // Fragments per ISA 16-bit layout: lane group g holds K = g*8+0..7
    // (elements 0..7) and K = 16+g*8+0..7 (elements 8..15).
    v16bf aF[4], bF[2];
#pragma unroll
    for (int i = 0; i < 4; ++i) {
      const __bf16* ap = &sA[cur][wm + i * 16 + r][0];
      aF[i] = cat16(*(const v8bf*)(ap + g * 8),
                    *(const v8bf*)(ap + 16 + g * 8));
    }
#pragma unroll
    for (int j = 0; j < 2; ++j) {
      const __bf16* bp = &sB[cur][wn + j * 16 + r][0];
      bF[j] = cat16(*(const v8bf*)(bp + g * 8),
                    *(const v8bf*)(bp + 16 + g * 8));
    }
#pragma unroll
    for (int i = 0; i < 4; ++i)
#pragma unroll
      for (int j = 0; j < 2; ++j)
        acc[i][j] = wmma_bf16(aF[i], bF[j], acc[i][j]);

    // Finish staging B into the other buffer (disjoint from cur reads)
    if (more) {
#pragma unroll
      for (int j = 0; j < 8; ++j)
        *(unsigned*)&sB[nxt][nb + j][kp] = pack2(b0[j], b1[j]);
    }
    wait_async();
    __syncthreads();
  }

  // C/D layout: element e, group g -> M = e + 8*g, N = r (within 16x16 tile)
#pragma unroll
  for (int i = 0; i < 4; ++i)
#pragma unroll
    for (int j = 0; j < 2; ++j)
#pragma unroll
      for (int e = 0; e < 8; ++e) {
        size_t row = (size_t)(m0 + wm + i * 16 + e + 8 * g);
        size_t col = (size_t)(n0 + wn + j * 16 + r);
        if (OUT_F32) Cf[row * N + col] = acc[i][j][e];
        else         Cb[row * N + col] = (__bf16)acc[i][j][e];
      }
}

// ---------------------------------------------------------------------------
// Flash attention, causal. Grid: (S/128, H, B). Block: 256 thr = 8 waves.
// Each wave owns 16 query rows; KV processed in 32-key blocks.
__global__ __launch_bounds__(256) void flash_attn_kernel(
    const __bf16* __restrict__ Q, const __bf16* __restrict__ Kmat,
    const __bf16* __restrict__ V, __bf16* __restrict__ O) {
  const int S = 2048, D = 2048, DK = 128;
  __shared__ __bf16 sK[32][128];    // [key][d]
  __shared__ __bf16 sVt[128][32];   // [d][key] (transposed V)
  __shared__ __bf16 sP[8][16][32];  // per-wave P staging (C->A relayout)

  const int tid  = threadIdx.x;
  const int lane = tid & 31;
  const int wave = tid >> 5;
  const int r = lane & 15;
  const int g = lane >> 4;
  const int qb = blockIdx.x;
  const int h  = blockIdx.y;
  const size_t bS = (size_t)blockIdx.z * S;
  const int qStart = qb * 128 + wave * 16;
  const float scale = 0.08838834764831845f;  // 1/sqrt(128)

  // Q fragments (A layout), loaded once: 4 K-steps over dk=128
  v16bf qF[4];
  {
    const __bf16* qrow = Q + (bS + qStart + r) * D + h * DK;
#pragma unroll
    for (int d = 0; d < 4; ++d)
      qF[d] = cat16(*(const v8bf*)(qrow + d * 32 + g * 8),
                    *(const v8bf*)(qrow + d * 32 + 16 + g * 8));
  }

  v8f acc[8] = {};     // O accumulator: 16 queries x 128 dk
  float m_e[8], l_e[8];
#pragma unroll
  for (int e = 0; e < 8; ++e) { m_e[e] = -3.0e38f; l_e[e] = 0.0f; }

  const int kvEnd = qb * 128 + 128;  // causal: keys <= last query in block
  for (int kb = 0; kb < kvEnd; kb += 32) {
    // Stage K (row-major) via async DMA: 32 keys x 128 dk
#pragma unroll
    for (int it = 0; it < 2; ++it) {
      int idx = tid + it * 256;     // 0..511
      int key = idx >> 4;           // 0..31
      int c   = (idx & 15) * 8;     // 0..120
      async_copy_b128(Kmat + (bS + kb + key) * D + h * DK + c, &sK[key][c]);
    }
    // Stage V transposed with packed b32 stores: key pairs
    {
      int k2 = (tid >> 4) * 2;      // even key: 0..30
      int c  = (tid & 15) * 8;      // 0..120
      const __bf16* vp = V + (bS + kb + k2) * D + h * DK + c;
      v8bf v0 = *(const v8bf*)vp;
      v8bf v1 = *(const v8bf*)(vp + D);
#pragma unroll
      for (int j = 0; j < 8; ++j)
        *(unsigned*)&sVt[c + j][k2] = pack2(v0[j], v1[j]);
    }
    wait_async();
    __syncthreads();

    // Prefetch next K/V tile into cache while this tile computes
    if (kb + 32 < kvEnd) {
      size_t prow = (bS + kb + 32 + (tid >> 3)) * D + h * DK + (tid & 7) * 16;
      __builtin_prefetch(Kmat + prow, 0, 0);
      __builtin_prefetch(V + prow, 0, 0);
    }

    // S = Q * K^T : two 16-key subtiles, 4 contraction steps each
    v8f sc[2];
#pragma unroll
    for (int nt = 0; nt < 2; ++nt) {
      v8f s = {};
#pragma unroll
      for (int d = 0; d < 4; ++d) {
        const __bf16* kp = &sK[nt * 16 + r][d * 32];
        v16bf kF = cat16(*(const v8bf*)(kp + g * 8),
                         *(const v8bf*)(kp + 16 + g * 8));
        s = wmma_bf16(qF[d], kF, s);
      }
      sc[nt] = s;
    }

    // Scale + causal mask (C layout: row = qStart+e+8g, key = kb+nt*16+r)
    float p0[8], p1[8], rm[8], rs[8];
#pragma unroll
    for (int e = 0; e < 8; ++e) {
      int qg = qStart + e + 8 * g;
      p0[e] = (kb + r      <= qg) ? sc[0][e] * scale : -3.0e38f;
      p1[e] = (kb + 16 + r <= qg) ? sc[1][e] * scale : -3.0e38f;
      rm[e] = fmaxf(p0[e], p1[e]);
    }
    // Row max across the 16 lanes of each half-group
#pragma unroll
    for (int off = 1; off < 16; off <<= 1)
#pragma unroll
      for (int e = 0; e < 8; ++e)
        rm[e] = fmaxf(rm[e], __shfl_xor(rm[e], off, 32));

#pragma unroll
    for (int e = 0; e < 8; ++e) {
      float nm    = fmaxf(m_e[e], rm[e]);
      float alpha = __expf(m_e[e] - nm);
      p0[e] = __expf(p0[e] - nm);
      p1[e] = __expf(p1[e] - nm);
      rs[e] = p0[e] + p1[e];
      m_e[e] = nm;
      l_e[e] *= alpha;
#pragma unroll
      for (int f = 0; f < 8; ++f) acc[f][e] *= alpha;
    }
#pragma unroll
    for (int off = 1; off < 16; off <<= 1)
#pragma unroll
      for (int e = 0; e < 8; ++e)
        rs[e] += __shfl_xor(rs[e], off, 32);
#pragma unroll
    for (int e = 0; e < 8; ++e) l_e[e] += rs[e];

    // Relayout P (C layout -> A layout) via per-wave LDS
#pragma unroll
    for (int e = 0; e < 8; ++e) {
      sP[wave][e + 8 * g][r]      = (__bf16)p0[e];
      sP[wave][e + 8 * g][16 + r] = (__bf16)p1[e];
    }
    asm volatile("" ::: "memory");  // keep LDS write->read order (same wave)
    const __bf16* pp = &sP[wave][r][0];
    v16bf pF = cat16(*(const v8bf*)(pp + g * 8),
                     *(const v8bf*)(pp + 16 + g * 8));

    // O += P * V : 8 dk subtiles, K-dim = 32 keys (one WMMA each)
#pragma unroll
    for (int f = 0; f < 8; ++f) {
      const __bf16* vp = &sVt[f * 16 + r][0];
      v16bf vF = cat16(*(const v8bf*)(vp + g * 8),
                       *(const v8bf*)(vp + 16 + g * 8));
      acc[f] = wmma_bf16(pF, vF, acc[f]);
    }
    __syncthreads();
  }

  // Normalize and store O (bf16, feeds final GEMM)
#pragma unroll
  for (int e = 0; e < 8; ++e) {
    float inv = 1.0f / l_e[e];
    __bf16* orow = O + (bS + qStart + e + 8 * g) * D + h * DK;
#pragma unroll
    for (int f = 0; f < 8; ++f)
      orow[f * 16 + r] = (__bf16)(acc[f][e] * inv);
  }
}

// ---------------------------------------------------------------------------
extern "C" void kernel_launch(void* const* d_in, const int* in_sizes, int n_in,
                              void* d_out, int out_size, void* d_ws, size_t ws_size,
                              hipStream_t stream) {
  const int B = 2, S = 2048, D = 2048;
  const int M  = B * S;          // 4096
  const int nX = M * D;          // 8388608
  const int nW = D * D;          // 4194304

  const float* x  = (const float*)d_in[0];
  const float* Wq = (const float*)d_in[1];
  const float* Wk = (const float*)d_in[2];
  const float* Wv = (const float*)d_in[3];
  const float* Wo = (const float*)d_in[4];

  // Workspace carve (bf16): x, 4 weights, Q, K, V, O  => ~117 MB
  __bf16* ws  = (__bf16*)d_ws;
  __bf16* xb  = ws;
  __bf16* wqb = xb  + nX;
  __bf16* wkb = wqb + nW;
  __bf16* wvb = wkb + nW;
  __bf16* wob = wvb + nW;
  __bf16* Qb  = wob + nW;
  __bf16* Kb  = Qb  + nX;
  __bf16* Vb  = Kb  + nX;
  __bf16* Ob  = Vb  + nX;

  f32_to_bf16_kernel<<<1024, 256, 0, stream>>>((const float4*)x,  xb,  nX / 4);
  f32_to_bf16_kernel<<<512,  256, 0, stream>>>((const float4*)Wq, wqb, nW / 4);
  f32_to_bf16_kernel<<<512,  256, 0, stream>>>((const float4*)Wk, wkb, nW / 4);
  f32_to_bf16_kernel<<<512,  256, 0, stream>>>((const float4*)Wv, wvb, nW / 4);
  f32_to_bf16_kernel<<<512,  256, 0, stream>>>((const float4*)Wo, wob, nW / 4);

  dim3 gg(D / 128, M / 128);  // (16, 32)
  gemm_bf16_kernel<false><<<gg, 256, 0, stream>>>(xb, wqb, nullptr, Qb, M, D, D);
  gemm_bf16_kernel<false><<<gg, 256, 0, stream>>>(xb, wkb, nullptr, Kb, M, D, D);
  gemm_bf16_kernel<false><<<gg, 256, 0, stream>>>(xb, wvb, nullptr, Vb, M, D, D);

  flash_attn_kernel<<<dim3(S / 128, 16, B), 256, 0, stream>>>(Qb, Kb, Vb, Ob);

  gemm_bf16_kernel<true><<<gg, 256, 0, stream>>>(Ob, wob, (float*)d_out, nullptr, M, D, D);
}
